// BGATLayer_19404662243973
// MI455X (gfx1250) — matrix operand order repaired and checked by reading, not compiled, not run on hardware
//
#include <hip/hip_runtime.h>

#define ALPHA 0.2f
#define B_DIM 64
#define N_DIM 1024
#define C_DIM 512
#define F_DIM 512

typedef __attribute__((ext_vector_type(16))) __bf16 v16bf;
typedef __attribute__((ext_vector_type(8)))  __bf16 v8bf;
typedef __attribute__((ext_vector_type(8)))  float  v8f;
typedef __attribute__((ext_vector_type(4)))  float  v4f;

__device__ __forceinline__ v8f zero8() {
    v8f z;
#pragma unroll
    for (int i = 0; i < 8; ++i) z[i] = 0.0f;
    return z;
}

// ---------------------------------------------------------------- prep: x -> bf16
__global__ __launch_bounds__(256) void prep_x(const float* __restrict__ x,
                                              __bf16* __restrict__ xb) {
    size_t base = ((size_t)blockIdx.x * 256 + threadIdx.x) * 8;
    v4f a = *(const v4f*)(x + base);
    v4f b = *(const v4f*)(x + base + 4);
    v8bf o;
#pragma unroll
    for (int i = 0; i < 4; ++i) { o[i] = (__bf16)a[i]; o[4 + i] = (__bf16)b[i]; }
    *(v8bf*)(xb + base) = o;
}

// ---------------------------------------------------------------- prep: W -> Wt bf16 (F x C)
__global__ __launch_bounds__(256) void prep_w(const float* __restrict__ W,
                                              __bf16* __restrict__ wt) {
    int idx = blockIdx.x * 256 + threadIdx.x;      // 262144 elems
    int f = idx >> 9, c = idx & 511;
    wt[idx] = (__bf16)W[c * F_DIM + f];
}

// ---------------------------------------------------------------- GEMM1: h = x @ W (bf16 WMMA)
// grid 4096 blocks of 256. Block = one 16-row m-tile across all 32 f-tiles.
// A panel (16x512 bf16, contiguous 16KB) staged once in LDS, shared by 8 waves x 4 f-tiles.
#define A_STRIDE 520   // bf16 elems; 1040 B row stride (16B multiple, de-patterned banks)
__global__ __launch_bounds__(256) void gemm1(const __bf16* __restrict__ xb,
                                             const __bf16* __restrict__ wt,
                                             __bf16* __restrict__ hb,    // row-major (B*N, F)
                                             __bf16* __restrict__ hbt) { // transposed (B, F, N)
    __shared__ __bf16 atile[16 * A_STRIDE];
    int mtile = blockIdx.x;                            // 0..4095
    int tid = threadIdx.x;

    // cooperative, fully coalesced A-panel load: 8192 bf16 = 16KB contiguous
    const __bf16* asrc = xb + (size_t)mtile * 16 * C_DIM;
    {
        int e0 = tid * 32;
#pragma unroll
        for (int c = 0; c < 4; ++c) {
            int e = e0 + c * 8;
            int r = e >> 9, col = e & 511;
            *(v8bf*)(&atile[r * A_STRIDE + col]) = *(const v8bf*)(asrc + e);
        }
    }
    __syncthreads();

    int wv = tid >> 5, lane = tid & 31;
    int arow  = lane & 15;
    int kbase = (lane >> 4) * 8;
    int koff  = (lane >> 4) * 16;
    const __bf16* albase = &atile[arow * A_STRIDE + kbase];

    v8f acc[4];
#pragma unroll
    for (int nt = 0; nt < 4; ++nt) acc[nt] = zero8();

    for (int k = 0; k < C_DIM; k += 32) {
        v16bf af;
        ((v8bf*)&af)[0] = *(const v8bf*)(albase + k);
        ((v8bf*)&af)[1] = *(const v8bf*)(albase + k + 16);
#pragma unroll
        for (int nt = 0; nt < 4; ++nt) {
            int ftile = wv * 4 + nt;
            const __bf16* bp = wt + (size_t)(ftile * 16 + (lane & 15)) * C_DIM + k + kbase;
            v16bf bf_;
            ((v8bf*)&bf_)[0] = *(const v8bf*)bp;
            ((v8bf*)&bf_)[1] = *(const v8bf*)(bp + 16);
            acc[nt] = __builtin_amdgcn_wmma_f32_16x16x32_bf16(false, af, false, bf_,
                                                              (short)0, acc[nt], false, false);
        }
    }

    // C layout: VGPR v, lane l -> row = v + 8*(l/16), col = l%16
    int crow0 = mtile * 16 + (lane >> 4) * 8;
    int bb = mtile >> 6;                               // 64 m-tiles per batch
    int n0 = ((mtile & 63) << 4) + ((lane >> 4) << 3);
#pragma unroll
    for (int nt = 0; nt < 4; ++nt) {
        int ccol = (wv * 4 + nt) * 16 + (lane & 15);
        v8bf packed;
#pragma unroll
        for (int v = 0; v < 8; ++v) {
            __bf16 hv = (__bf16)acc[nt][v];
            hb[(size_t)(crow0 + v) * F_DIM + ccol] = hv;
            packed[v] = hv;
        }
        *(v8bf*)(hbt + ((size_t)bb * F_DIM + ccol) * N_DIM + n0) = packed;
    }
}

// ---------------------------------------------------------------- s1/s2 = h . a1/a2
__global__ __launch_bounds__(256) void skern(const __bf16* __restrict__ hb,
                                             const float* __restrict__ a,
                                             float* __restrict__ s1,
                                             float* __restrict__ s2) {
    int row  = blockIdx.x * 8 + (threadIdx.x >> 5);    // 65536 rows
    int lane = threadIdx.x & 31;
    const __bf16* hp = hb + (size_t)row * F_DIM + lane * 16;
    float a1acc = 0.0f, a2acc = 0.0f;
#pragma unroll
    for (int t = 0; t < 16; ++t) {
        float hv = (float)hp[t];
        a1acc += hv * a[lane * 16 + t];
        a2acc += hv * a[F_DIM + lane * 16 + t];
    }
#pragma unroll
    for (int m = 16; m >= 1; m >>= 1) {
        a1acc += __shfl_xor(a1acc, m, 32);
        a2acc += __shfl_xor(a2acc, m, 32);
    }
    if (lane == 0) { s1[row] = a1acc; s2[row] = a2acc; }
}

// ---------------------------------------------------------------- attention + h' = att@h + epilogue
// grid (64 i-tiles, 64 batches), 256 threads. Two-pass softmax over rank-1 scores,
// then stream j in super-chunks of 128: p-tile computed cooperatively ONCE into LDS,
// all 8 waves read A-fragments from LDS and run bf16 WMMA.
#define P_STRIDE 136   // bf16 elems; 272 B row stride
__global__ __launch_bounds__(256) void attn(const __bf16* __restrict__ hbt,
                                            const float* __restrict__ s1g,
                                            const float* __restrict__ s2g,
                                            const float* __restrict__ mask,
                                            const float* __restrict__ beta,
                                            float* __restrict__ out) {
    __shared__ float s2_sh[N_DIM];
    __shared__ float s1_sh[16], m_sh[16], l_sh[16];
    __shared__ __bf16 p_sh[16 * P_STRIDE];

    int b = blockIdx.y;
    int itile = blockIdx.x;
    int tid = threadIdx.x;

    for (int j = tid; j < N_DIM; j += 256) s2_sh[j] = s2g[b * N_DIM + j];
    if (tid < 16) s1_sh[tid] = s1g[b * N_DIM + itile * 16 + tid];
    __syncthreads();

    // ---- pass A: per-row max then sum(exp) over j (16 threads per row)
    int row = tid >> 4, jl = tid & 15;
    const float* mrowp = mask + (size_t)(itile * 16 + row) * N_DIM;
    float s1v = s1_sh[row];
    float mx = -3.4e38f;
    for (int j = jl; j < N_DIM; j += 16) {
        float e = s1v + s2_sh[j];
        e = (e > 0.0f ? e : ALPHA * e) * mrowp[j];
        mx = fmaxf(mx, e);
    }
#pragma unroll
    for (int m = 8; m >= 1; m >>= 1) mx = fmaxf(mx, __shfl_xor(mx, m, 32));
    if (jl == 0) m_sh[row] = mx;
    __syncthreads();

    float mrow = m_sh[row], ssum = 0.0f;
    for (int j = jl; j < N_DIM; j += 16) {
        float e = s1v + s2_sh[j];
        e = (e > 0.0f ? e : ALPHA * e) * mrowp[j];
        ssum += __expf(e - mrow);
    }
#pragma unroll
    for (int m = 8; m >= 1; m >>= 1) ssum += __shfl_xor(ssum, m, 32);
    if (jl == 0) l_sh[row] = ssum;
    __syncthreads();

    // ---- pass B
    int wv = tid >> 5, lane = tid & 31;
    // p-producer role: thread t -> row t/16, j-range (t%16)*8 .. +8 within super-chunk
    int prow = tid >> 4;
    int pj   = (tid & 15) * 8;
    const float* maskP = mask + (size_t)(itile * 16 + prow) * N_DIM;
    float s1p = s1_sh[prow], mp = m_sh[prow], lip = 1.0f / l_sh[prow];

    // WMMA consumer role
    int arow = lane & 15;
    int kg   = (lane >> 4) * 8;
    int koff = (lane >> 4) * 16;
    const __bf16* hbase = hbt + (size_t)b * F_DIM * N_DIM;
    int ncol = wv * 64 + (lane & 15);

    v8f acc[4];
#pragma unroll
    for (int nt = 0; nt < 4; ++nt) acc[nt] = zero8();

    for (int jc = 0; jc < N_DIM; jc += 128) {
        __builtin_prefetch(maskP + jc + 128 + pj, 0, 0);   // global_prefetch next mask super-tile
        // cooperative p-tile (16x128) into LDS: 8 exp per thread
        {
            v4f mk0 = *(const v4f*)(maskP + jc + pj);
            v4f mk1 = *(const v4f*)(maskP + jc + pj + 4);
            v8bf pv;
#pragma unroll
            for (int t = 0; t < 8; ++t) {
                float e = s1p + s2_sh[jc + pj + t];
                float mk = (t < 4) ? mk0[t] : mk1[t - 4];
                e = (e > 0.0f ? e : ALPHA * e) * mk;
                pv[t] = (__bf16)(__expf(e - mp) * lip);
            }
            *(v8bf*)(&p_sh[prow * P_STRIDE + pj]) = pv;
        }
        __syncthreads();
#pragma unroll
        for (int ks = 0; ks < 4; ++ks) {
            v16bf af;
            const __bf16* ap = &p_sh[arow * P_STRIDE + ks * 32 + kg];
            ((v8bf*)&af)[0] = *(const v8bf*)ap;
            ((v8bf*)&af)[1] = *(const v8bf*)(ap + 16);
#pragma unroll
            for (int nt = 0; nt < 4; ++nt) {
                const __bf16* bp = hbase + (size_t)(ncol + nt * 16) * N_DIM + jc + ks * 32 + kg;
                v16bf bf_;
                ((v8bf*)&bf_)[0] = *(const v8bf*)bp;
                ((v8bf*)&bf_)[1] = *(const v8bf*)(bp + 16);
                acc[nt] = __builtin_amdgcn_wmma_f32_16x16x32_bf16(false, af, false, bf_,
                                                                  (short)0, acc[nt], false, false);
            }
        }
        __syncthreads();
    }

    // ---- epilogue: out = elu(h' + beta*h)
    float betav = beta[0];
    int r0 = (lane >> 4) * 8;
#pragma unroll
    for (int nt = 0; nt < 4; ++nt) {
        int f = ncol + nt * 16;
        v8bf hres = *(const v8bf*)(hbase + (size_t)f * N_DIM + itile * 16 + r0);
#pragma unroll
        for (int v = 0; v < 8; ++v) {
            float val = acc[nt][v] + betav * (float)hres[v];
            val = val > 0.0f ? val : (__expf(val) - 1.0f);
            out[((size_t)b * N_DIM + itile * 16 + r0 + v) * F_DIM + f] = val;
        }
    }
}

// ---------------------------------------------------------------- launch
extern "C" void kernel_launch(void* const* d_in, const int* in_sizes, int n_in,
                              void* d_out, int out_size, void* d_ws, size_t ws_size,
                              hipStream_t stream) {
    (void)in_sizes; (void)n_in; (void)out_size; (void)ws_size;
    const float* x    = (const float*)d_in[0];
    const float* W    = (const float*)d_in[1];
    const float* a    = (const float*)d_in[2];
    const float* beta = (const float*)d_in[3];
    const float* mask = (const float*)d_in[4];
    float* out = (float*)d_out;

    char* ws = (char*)d_ws;
    const size_t SZ_XB  = (size_t)B_DIM * N_DIM * C_DIM * 2;   // 64 MB
    const size_t SZ_WT  = (size_t)C_DIM * F_DIM * 2;           // 512 KB
    const size_t SZ_HB  = (size_t)B_DIM * N_DIM * F_DIM * 2;   // 64 MB
    const size_t SZ_HBT = SZ_HB;                               // 64 MB
    __bf16* xb  = (__bf16*)(ws);
    __bf16* wt  = (__bf16*)(ws + SZ_XB);
    __bf16* hb  = (__bf16*)(ws + SZ_XB + SZ_WT);
    __bf16* hbt = (__bf16*)(ws + SZ_XB + SZ_WT + SZ_HB);
    float*  s1  = (float*)(ws + SZ_XB + SZ_WT + SZ_HB + SZ_HBT);
    float*  s2  = s1 + (size_t)B_DIM * N_DIM;

    prep_x<<<16384, 256, 0, stream>>>(x, xb);                  // 33554432 / (256*8)
    prep_w<<<1024, 256, 0, stream>>>(W, wt);
    gemm1<<<4096, 256, 0, stream>>>(xb, wt, hb, hbt);
    skern<<<8192, 256, 0, stream>>>(hb, a, s1, s2);
    attn<<<dim3(64, 64), 256, 0, stream>>>(hbt, s1, s2, mask, beta, out);
}
